// SocialRecommenderSimplified_87866440942251
// MI455X (gfx1250) — matrix operand order repaired and checked by reading, not compiled, not run on hardware
//
#include <hip/hip_runtime.h>
#include <hip/hip_bf16.h>

typedef __attribute__((ext_vector_type(16))) _Float16 v16h;
typedef __attribute__((ext_vector_type(8)))  float    v8f;

#define NU 100000
#define NI 50000
#define DD 64
#define NNZ_UI 1600000
#define NNZ_S  800000

// ---------------------------------------------------------------- zero
__global__ void zero_kernel(float* __restrict__ p, int n) {
    int i = blockIdx.x * blockDim.x + threadIdx.x;
    if (i < n) p[i] = 0.0f;
}

// ---------------------------------------------------------------- scatter-add SpMM
// agg[rows[e], :] += vals[e] * X[cols[e], :]
// one edge per wave32; each lane handles 2 consecutive floats (float2)
__global__ __launch_bounds__(256) void scatter_kernel(
    const int* __restrict__ rows, const int* __restrict__ cols,
    const float* __restrict__ vals, const float* __restrict__ X,
    float* __restrict__ agg, int nnz)
{
    int tid  = blockIdx.x * blockDim.x + threadIdx.x;
    int e    = tid >> 5;
    if (e >= nnz) return;
    int lane = tid & 31;
    int r    = rows[e];
    int c    = cols[e];
    float v  = vals[e];
    const float2 s = ((const float2*)(X + (size_t)c * DD))[lane];
    float* dst = agg + (size_t)r * DD + lane * 2;
    unsafeAtomicAdd(dst,     v * s.x);   // global_atomic_add_f32
    unsafeAtomicAdd(dst + 1, v * s.y);
}

// ---------------------------------------------------------------- fused GEMM(+bias)+residual+LayerNorm
// out = LayerNorm(xin + Xagg @ Wl + bl) * g + bb      rows in 16-row tiles, one wave per tile
__global__ __launch_bounds__(256) void gemm_ln_kernel(
    const float* __restrict__ Xagg,  // [N,64]  A matrix
    const float* __restrict__ Wl,    // [64,64] B matrix (row-major: W[k*64+n])
    const float* __restrict__ bl,    // [64]
    const float* __restrict__ xin,   // [N,64]  residual input
    const float* __restrict__ g,     // [64] ln gamma
    const float* __restrict__ bb,    // [64] ln beta
    float* __restrict__ out,         // [N,64]
    int ntiles)
{
    int wave = (blockIdx.x * blockDim.x + threadIdx.x) >> 5;
    if (wave >= ntiles) return;              // wave-uniform -> EXEC stays all-ones
    int lane = threadIdx.x & 31;
    int m    = lane & 15;
    int half = lane >> 4;

    // ---- A operands: 16x32 f16, lane holds row m; VGPR v: K = 2*(v&3) + 8*half + 16*(v>>2)
    const float* arow = Xagg + (size_t)(wave * 16 + m) * DD;
    v16h a0, a1;
#pragma unroll
    for (int v = 0; v < 8; ++v) {
        int k0 = ((v & 3) << 1) + (half << 3) + ((v >> 2) << 4);
        a0[2*v]   = (_Float16)arow[k0];
        a0[2*v+1] = (_Float16)arow[k0 + 1];
        a1[2*v]   = (_Float16)arow[k0 + 32];
        a1[2*v+1] = (_Float16)arow[k0 + 33];
    }

    // ---- 4 column tiles of 16: B operands 32x16 f16, lane holds col n; VGPR v: K = 2*(v&3) + 8*half + 16*(v>>2)
    v8f c[4];
#pragma unroll
    for (int ct = 0; ct < 4; ++ct) {
        int n = ct * 16 + m;
        v16h b0, b1;
#pragma unroll
        for (int v = 0; v < 8; ++v) {
            int k = ((v & 3) << 1) + (half << 3) + ((v >> 2) << 4);
            b0[2*v]   = (_Float16)Wl[(k)      * DD + n];
            b0[2*v+1] = (_Float16)Wl[(k + 1)  * DD + n];
            b1[2*v]   = (_Float16)Wl[(k + 32) * DD + n];
            b1[2*v+1] = (_Float16)Wl[(k + 33) * DD + n];
        }
        v8f z = {};
        z = __builtin_amdgcn_wmma_f32_16x16x32_f16(false, a0, false, b0, (short)0, z, false, false);
        z = __builtin_amdgcn_wmma_f32_16x16x32_f16(false, a1, false, b1, (short)0, z, false, false);
        c[ct] = z;
    }

    // ---- epilogue: t = xin + gemm + bias ; C/D layout: VGPR r -> row r+8*half, lane -> col ct*16+m
    float t[8][4];
#pragma unroll
    for (int r = 0; r < 8; ++r) {
        int row = wave * 16 + r + half * 8;
#pragma unroll
        for (int ct = 0; ct < 4; ++ct) {
            int col = ct * 16 + m;
            t[r][ct] = xin[(size_t)row * DD + col] + c[ct][r] + bl[col];
        }
    }
#pragma unroll
    for (int r = 0; r < 8; ++r) {
        float s = t[r][0] + t[r][1] + t[r][2] + t[r][3];
        float q = t[r][0]*t[r][0] + t[r][1]*t[r][1] + t[r][2]*t[r][2] + t[r][3]*t[r][3];
#pragma unroll
        for (int off = 1; off < 16; off <<= 1) {   // reduce within each 16-lane half-group
            s += __shfl_xor(s, off, 32);
            q += __shfl_xor(q, off, 32);
        }
        float mean = s * (1.0f / 64.0f);
        float var  = q * (1.0f / 64.0f) - mean * mean;
        float rstd = rsqrtf(var + 1e-5f);
        int row = wave * 16 + r + half * 8;
#pragma unroll
        for (int ct = 0; ct < 4; ++ct) {
            int col = ct * 16 + m;
            out[(size_t)row * DD + col] = (t[r][ct] - mean) * rstd * g[col] + bb[col];
        }
    }
}

// ---------------------------------------------------------------- driver
extern "C" void kernel_launch(void* const* d_in, const int* in_sizes, int n_in,
                              void* d_out, int out_size, void* d_ws, size_t ws_size,
                              hipStream_t stream) {
    const float* user_emb = (const float*)d_in[0];
    const float* item_emb = (const float*)d_in[1];
    const float* W        = (const float*)d_in[2];   // [2,64,64]
    const float* b        = (const float*)d_in[3];   // [2,64]
    const float* ln_g     = (const float*)d_in[4];
    const float* ln_b     = (const float*)d_in[5];
    const float* ui_vals  = (const float*)d_in[6];
    const float* iu_vals  = (const float*)d_in[7];
    const float* s_vals   = (const float*)d_in[8];
    const int*   ui_rows  = (const int*)d_in[9];
    const int*   ui_cols  = (const int*)d_in[10];
    const int*   s_rows   = (const int*)d_in[11];
    const int*   s_cols   = (const int*)d_in[12];

    float* out    = (float*)d_out;
    float* u_out  = out;
    float* i_out  = out + (size_t)NU * DD;
    float* us_out = out + (size_t)(NU + NI) * DD;

    char* ws = (char*)d_ws;
    float* aggU = (float*)(ws);                                   // NU*64
    float* aggI = (float*)(ws + (size_t)NU * DD * 4);             // NI*64
    float* u1   = (float*)(ws + (size_t)(NU + NI) * DD * 4);      // NU*64
    float* i1   = (float*)(ws + (size_t)(2*NU + NI) * DD * 4);    // NI*64

    auto zero = [&](float* p, int n) {
        zero_kernel<<<(n + 255) / 256, 256, 0, stream>>>(p, n);
    };
    auto scat = [&](const int* r, const int* c, const float* v, const float* X,
                    float* agg, int nnz) {
        scatter_kernel<<<(nnz + 7) / 8, 256, 0, stream>>>(r, c, v, X, agg, nnz);
    };
    auto gln = [&](const float* Xa, const float* Wl, const float* bl,
                   const float* xin, float* o, int nrows) {
        int nt = nrows / 16;
        gemm_ln_kernel<<<(nt + 7) / 8, 256, 0, stream>>>(Xa, Wl, bl, xin, ln_g, ln_b, o, nt);
    };

    // ---- UI layer 0 (both scatters read pre-update embeddings)
    zero(aggU, NU * DD);  zero(aggI, NI * DD);
    scat(ui_rows, ui_cols, ui_vals, item_emb, aggU, NNZ_UI);
    scat(ui_cols, ui_rows, iu_vals, user_emb, aggI, NNZ_UI);
    gln(aggU, W,        b,      user_emb, u1, NU);
    gln(aggI, W,        b,      item_emb, i1, NI);
    // ---- UI layer 1
    zero(aggU, NU * DD);  zero(aggI, NI * DD);
    scat(ui_rows, ui_cols, ui_vals, i1, aggU, NNZ_UI);
    scat(ui_cols, ui_rows, iu_vals, u1, aggI, NNZ_UI);
    gln(aggU, W + 4096, b + 64, u1, u_out, NU);
    gln(aggI, W + 4096, b + 64, i1, i_out, NI);
    // ---- Social layer 0 (restarts from user_emb; reuse aggU and u1)
    zero(aggU, NU * DD);
    scat(s_rows, s_cols, s_vals, user_emb, aggU, NNZ_S);
    gln(aggU, W,        b,      user_emb, u1, NU);
    // ---- Social layer 1
    zero(aggU, NU * DD);
    scat(s_rows, s_cols, s_vals, u1, aggU, NNZ_S);
    gln(aggU, W + 4096, b + 64, u1, us_out, NU);
}
